// StructureMem_74603581932085
// MI455X (gfx1250) — matrix operand order repaired and checked by reading, not compiled, not run on hardware
//
#include <hip/hip_runtime.h>
#include <hip/hip_bf16.h>
#include <math.h>

typedef __attribute__((ext_vector_type(16))) _Float16 v16h;
typedef __attribute__((ext_vector_type(8)))  _Float16 v8h;
typedef __attribute__((ext_vector_type(4)))  _Float16 v4h;
typedef __attribute__((ext_vector_type(8)))  float    v8f;

#define NB    64
#define NC    200
#define ND    2048
#define NK    32
#define NHW   196
#define NMAP  14
#define NT2   16      // padded n-tiles of 16 -> 256 cols (uniform wave work)
#define NPAD  256
#define LDP   24      // padded LDS fragment stride (halves): 48B, 8B-aligned quads
#define DCH   16      // chunks over D for heatmap partial pass
#define DPC   (ND / DCH)   // 128

// ---------------------------------------------------------------------------
// K1: softmax max-prob + argmax per sample.  pred_val = 1/sum(exp(s - smax)).
// ---------------------------------------------------------------------------
__global__ __launch_bounds__(256) void k_scores(const float* __restrict__ scores,
                                                float* __restrict__ pred_val,
                                                int* __restrict__ pred_pos) {
  const int b = blockIdx.x, t = threadIdx.x;
  __shared__ float rv[256];
  __shared__ int   ri[256];
  __shared__ float smax;
  __shared__ int   spos;
  float v = (t < NC) ? scores[b * NC + t] : -INFINITY;
  rv[t] = v; ri[t] = t;
  __syncthreads();
  for (int s = 128; s > 0; s >>= 1) {
    if (t < s) {
      if (rv[t + s] > rv[t] || (rv[t + s] == rv[t] && ri[t + s] < ri[t])) {
        rv[t] = rv[t + s]; ri[t] = ri[t + s];
      }
    }
    __syncthreads();
  }
  if (t == 0) { smax = rv[0]; spos = ri[0]; }
  __syncthreads();
  rv[t] = (t < NC) ? expf(scores[b * NC + t] - smax) : 0.0f;
  __syncthreads();
  for (int s = 128; s > 0; s >>= 1) {
    if (t < s) rv[t] += rv[t + s];
    __syncthreads();
  }
  if (t == 0) { pred_val[b] = 1.0f / rv[0]; pred_pos[b] = spos; }
}

// ---------------------------------------------------------------------------
// K2a: single pass over feat (103 MB): per (b,hw) partial sum of relu (heatmap)
// and sum of squares (candidate L2 norm) over a 128-deep d-chunk.
// ---------------------------------------------------------------------------
__global__ __launch_bounds__(256) void k_hm_partial(const float* __restrict__ feat,
                                                    float* __restrict__ hm_part,
                                                    float* __restrict__ sq_part) {
  const int b = blockIdx.x, c = blockIdx.y, t = threadIdx.x;
  if (t >= NHW) return;
  const float* base = feat + (size_t)b * ND * NHW + (size_t)c * DPC * NHW + t;
  float s = 0.0f, q = 0.0f;
  for (int d = 0; d < DPC; ++d) {
    float v = base[(size_t)d * NHW];
    s += fmaxf(v, 0.0f);
    q += v * v;
  }
  hm_part[((size_t)b * DCH + c) * NHW + t] = s;
  sq_part[((size_t)b * DCH + c) * NHW + t] = q;
}

// ---------------------------------------------------------------------------
// K2b: reduce chunks -> hm[b,hw]; normalize over H (axis=1 of [B,14,14]);
// cand_norm = sqrt(sum of squares); then top-32 selection of normalized hm.
// ---------------------------------------------------------------------------
__global__ __launch_bounds__(256) void k_hm_topk(const float* __restrict__ hm_part,
                                                 const float* __restrict__ sq_part,
                                                 float* __restrict__ cand_norm,
                                                 float* __restrict__ pick_val,
                                                 int* __restrict__ pick_pos) {
  const int b = blockIdx.x, t = threadIdx.x;
  __shared__ float hm[NHW];
  __shared__ float inv[NMAP];
  __shared__ float rv[256];
  __shared__ int   ri[256];
  if (t < NHW) {
    float s = 0.0f, q = 0.0f;
    for (int c = 0; c < DCH; ++c) {
      s += hm_part[((size_t)b * DCH + c) * NHW + t];
      q += sq_part[((size_t)b * DCH + c) * NHW + t];
    }
    hm[t] = s;
    cand_norm[b * NHW + t] = sqrtf(q);
  }
  __syncthreads();
  if (t < NMAP) {             // norm over H for column w = t
    float ss = 0.0f;
    for (int h = 0; h < NMAP; ++h) { float v = hm[h * NMAP + t]; ss += v * v; }
    inv[t] = 1.0f / fmaxf(sqrtf(ss), 1e-12f);
  }
  __syncthreads();
  if (t < NHW) hm[t] *= inv[t % NMAP];
  __syncthreads();
  for (int it = 0; it < NK; ++it) {
    rv[t] = (t < NHW) ? hm[t] : -INFINITY;
    ri[t] = t;
    __syncthreads();
    for (int s = 128; s > 0; s >>= 1) {
      if (t < s) {
        if (rv[t + s] > rv[t] || (rv[t + s] == rv[t] && ri[t + s] < ri[t])) {
          rv[t] = rv[t + s]; ri[t] = ri[t + s];
        }
      }
      __syncthreads();
    }
    if (t == 0) {
      pick_val[b * NK + it] = rv[0];
      pick_pos[b * NK + it] = ri[0];
      hm[ri[0]] = -INFINITY;
    }
    __syncthreads();
  }
}

// ---------------------------------------------------------------------------
// K3: per-(b,k) L2 norm of bank columns feat_bank[labels[b], :, k].
// ---------------------------------------------------------------------------
__global__ __launch_bounds__(256) void k_bank_norm(const float* __restrict__ feat_bank,
                                                   const int* __restrict__ labels,
                                                   float* __restrict__ bank_norm) {
  const int b = blockIdx.x, t = threadIdx.x;
  const int k = t & 31, part = t >> 5;     // 8 parts x 256 d
  const float* base = feat_bank + (size_t)labels[b] * ND * NK;
  float q = 0.0f;
  for (int d = part * 256; d < part * 256 + 256; ++d) {
    float v = base[(size_t)d * NK + k];
    q += v * v;
  }
  __shared__ float red[256];
  red[t] = q;
  __syncthreads();
  for (int s = 128; s >= 32; s >>= 1) {
    if (t < s) red[t] += red[t + s];
    __syncthreads();
  }
  if (t < NK) bank_norm[b * NK + t] = sqrtf(red[t]);
}

// ---------------------------------------------------------------------------
// K4a: branch masks + sequential-overwrite resolution (last updater per class).
// ---------------------------------------------------------------------------
__global__ __launch_bounds__(256) void k_masks(const int* __restrict__ labels,
                                               const float* __restrict__ bank_conf,
                                               const float* __restrict__ pred_val,
                                               const int* __restrict__ pred_pos,
                                               float* __restrict__ fwd_mask,
                                               int* __restrict__ class_src) {
  __shared__ int           s_lbl[NB];
  __shared__ unsigned char s_upd[NB];
  const int t = threadIdx.x;
  if (t < NB) {
    int   lbl = labels[t];
    float pv  = pred_val[t];
    bool  correct = (pred_pos[t] == lbl);
    float lc  = bank_conf[lbl];
    bool  update = correct && (pv - lc > 0.1f);
    bool  fwd = ((correct && (lc - pv > 0.1f)) || !correct) && (lc != 0.0f);
    s_lbl[t] = lbl;
    s_upd[t] = update ? 1 : 0;
    fwd_mask[t] = fwd ? 1.0f : 0.0f;
  }
  __syncthreads();
  if (t < NC) {
    int src = -1;
    for (int b = 0; b < NB; ++b)
      if (s_upd[b] && s_lbl[b] == t) src = b;   // last writer wins
    class_src[t] = src;
  }
}

// ---------------------------------------------------------------------------
// K4b: small scatter outputs: new_conf_tr, new_conf, new_pos.
// ---------------------------------------------------------------------------
__global__ __launch_bounds__(32) void k_small_outputs(const int* __restrict__ class_src,
                                                      const float* __restrict__ bank_conf,
                                                      const float* __restrict__ bank_conf_tr,
                                                      const int* __restrict__ bank_position,
                                                      const float* __restrict__ pred_val,
                                                      const float* __restrict__ pick_val,
                                                      const int* __restrict__ pick_pos,
                                                      float* __restrict__ out_conf_tr,
                                                      float* __restrict__ out_conf,
                                                      int* __restrict__ out_pos) {
  const int c = blockIdx.x, k = threadIdx.x;
  const int src = class_src[c];
  if (k == 0) out_conf[c] = (src >= 0) ? pred_val[src] : bank_conf[c];
  out_conf_tr[c * NK + k] = (src >= 0) ? pick_val[src * NK + k] : bank_conf_tr[c * NK + k];
  out_pos[c * NK + k]     = (src >= 0) ? pick_pos[src * NK + k] : bank_position[c * NK + k];
}

// ---------------------------------------------------------------------------
// K5: new_feat_bank = copy, with updated rows gathered from feat at pick_pos.
// ---------------------------------------------------------------------------
__global__ __launch_bounds__(256) void k_bank_update(const float* __restrict__ feat_bank,
                                                     const float* __restrict__ feat,
                                                     const int* __restrict__ class_src,
                                                     const int* __restrict__ pick_pos,
                                                     float* __restrict__ out_bank) {
  const size_t idx = (size_t)blockIdx.x * 256 + threadIdx.x; // (c*ND + d)*NK + k
  const int    k   = (int)(idx & (NK - 1));
  const size_t cd  = idx >> 5;
  const int    d   = (int)(cd & (ND - 1));
  const int    c   = (int)(cd >> 11);
  const int    src = class_src[c];
  float v;
  if (src >= 0) {
    int pos = pick_pos[src * NK + k];
    v = feat[((size_t)src * ND + d) * NHW + pos];
  } else {
    v = feat_bank[idx];
  }
  out_bank[idx] = v;
}

// ---------------------------------------------------------------------------
// K6: WMMA cosine-sim GEMM. One block per sample b; 8 wave32; C = 32k x 256n
// (n padded so every wave owns k-tile (w&1) and n-tiles 4*(w>>1)+j, j=0..3 —
// fully uniform, branch-free). LDS tiles are stored *pre-swizzled into WMMA
// fragment order* (ISA 7.12.2): staging packs 4 K-consecutive f16 per
// ds_store_b64; each lane then loads its whole fragment as 2x ds_load_b128.
// Epilogue scales raw dots by 1/(max(|bank_k|,eps)*max(|cand_n|,eps)).
// ---------------------------------------------------------------------------
__global__ __launch_bounds__(256) void k_sim_wmma(const float* __restrict__ feat,
                                                  const float* __restrict__ feat_bank,
                                                  const int* __restrict__ labels,
                                                  const float* __restrict__ bank_norm,
                                                  const float* __restrict__ cand_norm,
                                                  float* __restrict__ sim) {
  const int b    = blockIdx.x;
  const int tid  = threadIdx.x;
  const int w    = tid >> 5;        // wave 0..7
  const int lane = tid & 31;
  const int ktile = w & 1;
  const int m    = lane & 15;
  const int hi   = lane >> 4;       // half-wave select

  // Fragment-order LDS: [tile][lane][elem], elem stride padded to LDP=24
  __shared__ alignas(32) _Float16 Afrag[2][32][LDP];
  __shared__ alignas(32) _Float16 Bfrag[NT2][32][LDP];

  const float* fbrow = feat_bank + (size_t)labels[b] * ND * NK;
  const float* frow  = feat + (size_t)b * ND * NHW;

  v8f acc[4];
  for (int j = 0; j < 4; ++j) acc[j] = (v8f){};

  for (int dc = 0; dc < ND; dc += 32) {
    // ---- stage A (32d x 32k): one K-quad per thread, swizzled to frag order.
    // elem(dd) = (dd&7) + 8*(dd>=16); lane-half hi = (dd>>3)&1; lane = hi*16+k%16
    {
      int k   = tid & 31;
      int ddb = (tid >> 5) << 2;                 // 0,4,...,28
      v4h pk;
#pragma unroll
      for (int r = 0; r < 4; ++r)
        pk[r] = (_Float16)fbrow[(size_t)(dc + ddb + r) * NK + k];
      int ahi = (ddb >> 3) & 1;
      int eb  = (ddb & 7) + ((ddb >> 4) << 3);
      *(v4h*)&Afrag[k >> 4][ahi * 16 + (k & 15)][eb] = pk;
    }
    // ---- stage B (32d x 256n, cols>195 clamp-dup col 195): 8 K-quads/thread.
    // elem(dd) = (dd&7) + 8*(dd>=16); lane-half hi = (dd>>3)&1; lane = hi*16 + nn%16
#pragma unroll
    for (int i = 0; i < 8; ++i) {
      int q   = i * 256 + tid;
      int nn  = q & 255;
      int ddb = (q >> 8) << 2;                   // 0,4,...,28
      int ns  = nn < NHW ? nn : (NHW - 1);
      v4h pk;
#pragma unroll
      for (int r = 0; r < 4; ++r)
        pk[r] = (_Float16)frow[(size_t)(dc + ddb + r) * NHW + ns];
      *(v4h*)&Bfrag[nn >> 4][(((ddb >> 3) & 1) << 4) + (nn & 15)][(ddb & 7) + ((ddb >> 4) << 3)] = pk;
    }
    __syncthreads();

    // ---- fragments: contiguous 32B per lane -> 2x ds_load_b128 each
    v8h alo = *(const v8h*)&Afrag[ktile][lane][0];
    v8h ahi8 = *(const v8h*)&Afrag[ktile][lane][8];
    v16h a = __builtin_shufflevector(alo, ahi8, 0, 1, 2, 3, 4, 5, 6, 7,
                                     8, 9, 10, 11, 12, 13, 14, 15);
#pragma unroll
    for (int j = 0; j < 4; ++j) {
      int tn = ((w >> 1) << 2) + j;
      v8h blo = *(const v8h*)&Bfrag[tn][lane][0];
      v8h bhi = *(const v8h*)&Bfrag[tn][lane][8];
      v16h bf = __builtin_shufflevector(blo, bhi, 0, 1, 2, 3, 4, 5, 6, 7,
                                        8, 9, 10, 11, 12, 13, 14, 15);
      acc[j] = __builtin_amdgcn_wmma_f32_16x16x32_f16(false, a, false, bf,
                                                      (short)0, acc[j], false, false);
    }
    __syncthreads();
  }

  // ---- epilogue: cosine scaling + store (C: lane=N, VGPR r -> M=r(+8 hi))
#pragma unroll
  for (int j = 0; j < 4; ++j) {
    int tn = ((w >> 1) << 2) + j;
    int nn = tn * 16 + m;
    float cn = (nn < NHW) ? fmaxf(cand_norm[b * NHW + nn], 1e-8f) : 1.0f;
#pragma unroll
    for (int r = 0; r < 8; ++r) {
      int k = ktile * 16 + r + (hi << 3);
      float bn = fmaxf(bank_norm[b * NK + k], 1e-8f);
      sim[((size_t)b * NK + k) * NPAD + nn] = acc[j][r] / (bn * cn);
    }
  }
}

// ---------------------------------------------------------------------------
// K7: argmax over the 196 valid candidates per (b,k) row.
// ---------------------------------------------------------------------------
__global__ __launch_bounds__(64) void k_argmax(const float* __restrict__ sim,
                                               int* __restrict__ match_pos) {
  const int bk = blockIdx.x, t = threadIdx.x;
  const float* row = sim + (size_t)bk * NPAD;
  float best = -INFINITY; int bi = 0;
  for (int n = t; n < NHW; n += 64) {
    float v = row[n];
    if (v > best) { best = v; bi = n; }
  }
  __shared__ float rv[64];
  __shared__ int   ri[64];
  rv[t] = best; ri[t] = bi;
  __syncthreads();
  for (int s = 32; s > 0; s >>= 1) {
    if (t < s) {
      if (rv[t + s] > rv[t] || (rv[t + s] == rv[t] && ri[t + s] < ri[t])) {
        rv[t] = rv[t + s]; ri[t] = ri[t + s];
      }
    }
    __syncthreads();
  }
  if (t == 0) match_pos[bk] = ri[0];
}

// ---------------------------------------------------------------------------
// K8: aligned_feat[b,d,k] = feat[b,d,match_pos[b,k]] * forward_mask[b]
// ---------------------------------------------------------------------------
__global__ __launch_bounds__(256) void k_aligned(const float* __restrict__ feat,
                                                 const int* __restrict__ match_pos,
                                                 const float* __restrict__ fwd_mask,
                                                 float* __restrict__ out) {
  const size_t idx = (size_t)blockIdx.x * 256 + threadIdx.x; // (b*ND + d)*NK + k
  const int    k   = (int)(idx & (NK - 1));
  const size_t bd  = idx >> 5;
  const int    d   = (int)(bd & (ND - 1));
  const int    b   = (int)(bd >> 11);
  const int    pos = match_pos[b * NK + k];
  out[idx] = feat[((size_t)b * ND + d) * NHW + pos] * fwd_mask[b];
}

// ---------------------------------------------------------------------------
extern "C" void kernel_launch(void* const* d_in, const int* in_sizes, int n_in,
                              void* d_out, int out_size, void* d_ws, size_t ws_size,
                              hipStream_t stream) {
  const float* scores        = (const float*)d_in[0];
  const float* feat          = (const float*)d_in[1];
  const float* feat_bank     = (const float*)d_in[2];
  const float* bank_conf     = (const float*)d_in[3];
  const float* bank_conf_tr  = (const float*)d_in[4];
  const int*   labels        = (const int*)d_in[5];
  const int*   bank_position = (const int*)d_in[6];

  float* out = (float*)d_out;
  float* out_aligned = out;                                          // 64*2048*32
  float* out_bank    = out + (size_t)NB * ND * NK;                   // 200*2048*32
  float* out_conf_tr = out_bank + (size_t)NC * ND * NK;              // 200*32
  float* out_conf    = out_conf_tr + (size_t)NC * NK;                // 200
  int*   out_pos     = (int*)(out_conf + NC);                        // 200*32 (bit-cast)

  // workspace layout (floats)
  float* wsf = (float*)d_ws;
  size_t o = 0;
  float* ws_pred_val  = wsf + o; o += NB;
  int*   ws_pred_pos  = (int*)(wsf + o); o += NB;
  float* ws_hm_part   = wsf + o; o += (size_t)NB * DCH * NHW;
  float* ws_sq_part   = wsf + o; o += (size_t)NB * DCH * NHW;
  float* ws_cand_norm = wsf + o; o += (size_t)NB * NHW;
  float* ws_pick_val  = wsf + o; o += (size_t)NB * NK;
  int*   ws_pick_pos  = (int*)(wsf + o); o += (size_t)NB * NK;
  float* ws_fwd_mask  = wsf + o; o += NB;
  int*   ws_class_src = (int*)(wsf + o); o += NC;
  float* ws_bank_norm = wsf + o; o += (size_t)NB * NK;
  float* ws_sim       = wsf + o; o += (size_t)NB * NK * NPAD;
  int*   ws_match     = (int*)(wsf + o); o += (size_t)NB * NK;
  (void)ws_size; (void)in_sizes; (void)n_in; (void)out_size;

  k_scores<<<NB, 256, 0, stream>>>(scores, ws_pred_val, ws_pred_pos);
  k_hm_partial<<<dim3(NB, DCH), 256, 0, stream>>>(feat, ws_hm_part, ws_sq_part);
  k_hm_topk<<<NB, 256, 0, stream>>>(ws_hm_part, ws_sq_part, ws_cand_norm,
                                    ws_pick_val, ws_pick_pos);
  k_bank_norm<<<NB, 256, 0, stream>>>(feat_bank, labels, ws_bank_norm);
  k_masks<<<1, 256, 0, stream>>>(labels, bank_conf, ws_pred_val, ws_pred_pos,
                                 ws_fwd_mask, ws_class_src);
  k_small_outputs<<<NC, NK, 0, stream>>>(ws_class_src, bank_conf, bank_conf_tr,
                                         bank_position, ws_pred_val, ws_pick_val,
                                         ws_pick_pos, out_conf_tr, out_conf, out_pos);
  k_bank_update<<<(NC * ND * NK) / 256, 256, 0, stream>>>(feat_bank, feat,
                                                          ws_class_src, ws_pick_pos,
                                                          out_bank);
  k_sim_wmma<<<NB, 256, 0, stream>>>(feat, feat_bank, labels, ws_bank_norm,
                                     ws_cand_norm, ws_sim);
  k_argmax<<<NB * NK, 64, 0, stream>>>(ws_sim, ws_match);
  k_aligned<<<(NB * ND * NK) / 256, 256, 0, stream>>>(feat, ws_match, ws_fwd_mask,
                                                      out_aligned);
}
